// _FakeMoeBlock_27693949125163
// MI455X (gfx1250) — compile-verified, matrix operand order checked by reading, not run on hardware
//
#include <hip/hip_runtime.h>
#include <math.h>

// ---------------- problem constants ----------------
#define T_TOKENS 4096
#define H_DIM    1024
#define I_DIM    1408
#define E_NUM    8
#define TOPK     2
// per-expert segments padded to 32 rows: 8192 assignments + 8*32 slack
#define NROWS_MAX 8448
#define ROW_TILES (NROWS_MAX / 32)   // 264 (32-row super-tiles)

typedef __attribute__((ext_vector_type(16))) __bf16 v16bf;
typedef __attribute__((ext_vector_type(8)))  float  v8f;

union FragAB { v16bf v; unsigned int u[8]; };

__device__ __forceinline__ unsigned short f32_to_bf16(float f) {
  unsigned int u = __float_as_uint(f);
  u += 0x7FFFu + ((u >> 16) & 1u);   // round-to-nearest-even
  return (unsigned short)(u >> 16);
}

// ---------------- workspace layout (bytes) ----------------
constexpr size_t alignup(size_t x) { return (x + 255) & ~size_t(255); }
constexpr size_t OFF_COUNTS = 0;                                         // 8 ints
constexpr size_t OFF_CURSOR = 32;                                        // 8 ints (contiguous w/ counts)
constexpr size_t OFF_OFFS   = 64;                                        // 9 ints
constexpr size_t OFF_TOPI   = 256;                                       // T*2 ints
constexpr size_t OFF_TOPW   = alignup(OFF_TOPI + (size_t)T_TOKENS * 2 * 4);
constexpr size_t OFF_TOK    = alignup(OFF_TOPW + (size_t)T_TOKENS * 2 * 4);
constexpr size_t OFF_WROW   = alignup(OFF_TOK  + (size_t)NROWS_MAX * 4);
constexpr size_t OFF_EROW   = alignup(OFF_WROW + (size_t)NROWS_MAX * 4);
constexpr size_t OFF_XB     = alignup(OFF_EROW + (size_t)NROWS_MAX * 4);
constexpr size_t OFF_GUB    = alignup(OFF_XB  + (size_t)T_TOKENS * H_DIM * 2);
constexpr size_t OFF_DNB    = alignup(OFF_GUB + (size_t)E_NUM * 2 * I_DIM * H_DIM * 2);
constexpr size_t OFF_HB     = alignup(OFF_DNB + (size_t)E_NUM * H_DIM * I_DIM * 2);

// ---------------- init: zero output + atomic counters ----------------
__global__ void k_init(float* __restrict__ out, int n, int* __restrict__ cnt) {
  int i = blockIdx.x * blockDim.x + threadIdx.x;
  if (i < 16) cnt[i] = 0;                       // counts[8] + cursor[8]
  for (int j = i; j < n; j += gridDim.x * blockDim.x) out[j] = 0.f;
}

// ---------------- f32 -> bf16 conversion ----------------
__global__ void k_cvt(const float* __restrict__ src, unsigned short* __restrict__ dst, int n) {
  for (int i = blockIdx.x * blockDim.x + threadIdx.x; i < n; i += gridDim.x * blockDim.x)
    dst[i] = f32_to_bf16(src[i]);
}

// ---------------- router: f32 logits, softmax, top-2 ----------------
__global__ void __launch_bounds__(256) k_router(const float* __restrict__ x,
                                                const float* __restrict__ wr,
                                                int* __restrict__ counts,
                                                int* __restrict__ topi,
                                                float* __restrict__ topw) {
  int wave = threadIdx.x >> 5;
  int lane = threadIdx.x & 31;
  int t = blockIdx.x * 8 + wave;
  if (t >= T_TOKENS) return;
  const float* xr = x + (size_t)t * H_DIM;
  float xv[32];
  #pragma unroll
  for (int j = 0; j < 32; ++j) xv[j] = xr[j * 32 + lane];
  float logit[E_NUM];
  #pragma unroll
  for (int e = 0; e < E_NUM; ++e) {
    const float* w = wr + (size_t)e * H_DIM;
    float acc = 0.f;
    #pragma unroll
    for (int j = 0; j < 32; ++j) acc += xv[j] * w[j * 32 + lane];
    #pragma unroll
    for (int m = 16; m > 0; m >>= 1) acc += __shfl_xor(acc, m, 32);
    logit[e] = acc;                      // all lanes hold full sum
  }
  float mx = logit[0];
  #pragma unroll
  for (int e = 1; e < E_NUM; ++e) mx = fmaxf(mx, logit[e]);
  float pe[E_NUM]; float s = 0.f;
  #pragma unroll
  for (int e = 0; e < E_NUM; ++e) { pe[e] = __expf(logit[e] - mx); s += pe[e]; }
  float inv = 1.f / s;
  int i1 = 0; float p1 = pe[0];
  #pragma unroll
  for (int e = 1; e < E_NUM; ++e) if (pe[e] > p1) { p1 = pe[e]; i1 = e; }  // ties -> lowest idx
  int i2 = (i1 == 0) ? 1 : 0; float p2 = pe[i2];
  #pragma unroll
  for (int e = 0; e < E_NUM; ++e)
    if (e != i1 && e != i2 && pe[e] > p2) { p2 = pe[e]; i2 = e; }
  if (lane == 0) {
    topi[t * 2 + 0] = i1; topw[t * 2 + 0] = p1 * inv;
    topi[t * 2 + 1] = i2; topw[t * 2 + 1] = p2 * inv;
    atomicAdd(&counts[i1], 1);
    atomicAdd(&counts[i2], 1);
  }
}

// ---------------- exclusive scan with 32-row padding ----------------
__global__ void k_scan(const int* __restrict__ counts, int* __restrict__ offs,
                       int* __restrict__ cursor) {
  if (threadIdx.x == 0 && blockIdx.x == 0) {
    int acc = 0;
    for (int e = 0; e < E_NUM; ++e) {
      offs[e] = acc;
      cursor[e] = acc;
      acc += (counts[e] + 31) & ~31;
    }
    offs[E_NUM] = acc;
  }
}

// ---------------- default-fill all rows (pad rows: tok=0, w=0) ----------------
__global__ void k_fill(const int* __restrict__ offs, int* __restrict__ tok,
                       float* __restrict__ wrow, int* __restrict__ erow) {
  int idx = blockIdx.x * blockDim.x + threadIdx.x;
  if (idx >= NROWS_MAX) return;
  int e = 0;
  #pragma unroll
  for (int k = 1; k < E_NUM; ++k) if (idx >= offs[k]) e = k;
  tok[idx] = 0; wrow[idx] = 0.f; erow[idx] = e;
}

// ---------------- scatter real assignments into segments ----------------
__global__ void k_build(const int* __restrict__ topi, const float* __restrict__ topw,
                        int* __restrict__ cursor, int* __restrict__ tok,
                        float* __restrict__ wrow) {
  int t = blockIdx.x * blockDim.x + threadIdx.x;
  if (t >= T_TOKENS) return;
  #pragma unroll
  for (int k = 0; k < TOPK; ++k) {
    int e = topi[t * 2 + k];
    int pos = atomicAdd(&cursor[e], 1);
    tok[pos] = t;
    wrow[pos] = topw[t * 2 + k];
  }
}

// ---- fragment k-map (ISA 16-bit A/B layout): dword pair jj -> k offset ----
__device__ __forceinline__ int kmap(int jj, int kof) {
  return ((jj < 4) ? (2 * jj) : (2 * jj + 8)) + kof;
}

// ---------------- gate/up GEMM + fused SiLU (bf16 WMMA) ----------------
// 32 tokens x 16 I-dims per wave; 4 WMMAs (2 row-groups x {gate,up}) per k-step.
#define GU_CHUNK  256
#define GU_STRIDE 264
__global__ void __launch_bounds__(256) k_gu(
    const unsigned short* __restrict__ xb,
    const unsigned short* __restrict__ gub,
    const int* __restrict__ offs,
    const int* __restrict__ tok,
    const int* __restrict__ erow,
    unsigned short* __restrict__ hb) {
  __shared__ unsigned short sA[32 * GU_STRIDE];   // 16.9 KB
  int total = offs[E_NUM];
  int row_base = blockIdx.y * 32;
  if (row_base >= total) return;        // uniform per block
  int e = erow[row_base];
  int tid = threadIdx.x;
  int wave = tid >> 5, lane = tid & 31;
  int i_base = (blockIdx.x * 8 + wave) * 16;
  int m   = lane & 15;
  int n   = lane & 15;
  int kof = (lane & 16) ? 8 : 0;
  const unsigned short* bg = gub + ((size_t)e * (2 * I_DIM) + (i_base + n)) * H_DIM;
  const unsigned short* bu = gub + ((size_t)e * (2 * I_DIM) + (I_DIM + i_base + n)) * H_DIM;
  v8f accg0 = {}, accg1 = {}, accu0 = {}, accu1 = {};

  for (int cc = 0; cc < H_DIM / GU_CHUNK; ++cc) {         // 4 chunks of 256 H-cols
    __syncthreads();
    // stage 32 gathered x rows x 256 bf16 = 4096 dwords, 16/thread
    #pragma unroll
    for (int it = 0; it < 16; ++it) {
      int idx = it * 256 + tid;
      int r = idx >> 7;                 // / 128 dwords per row
      int c = idx & 127;
      int trow = tok[row_base + r];
      unsigned int d = *(const unsigned int*)&xb[(size_t)trow * H_DIM + cc * GU_CHUNK + c * 2];
      *(unsigned int*)&sA[r * GU_STRIDE + c * 2] = d;
    }
    __syncthreads();
    for (int kk = 0; kk < GU_CHUNK / 32; ++kk) {          // 8 k-steps per chunk
      FragAB a0, a1, b0, b1;
      int kb_l = kk * 32;
      int kb_g = cc * GU_CHUNK + kk * 32;
      #pragma unroll
      for (int jj = 0; jj < 8; ++jj) {
        int k0 = kmap(jj, kof);
        a0.u[jj] = *(const unsigned int*)&sA[m * GU_STRIDE + kb_l + k0];
        a1.u[jj] = *(const unsigned int*)&sA[(m + 16) * GU_STRIDE + kb_l + k0];
        b0.u[jj] = *(const unsigned int*)&bg[kb_g + k0];
        b1.u[jj] = *(const unsigned int*)&bu[kb_g + k0];
      }
      accg0 = __builtin_amdgcn_wmma_f32_16x16x32_bf16(false, a0.v, false, b0.v, (short)0, accg0, false, false);
      accg1 = __builtin_amdgcn_wmma_f32_16x16x32_bf16(false, a1.v, false, b0.v, (short)0, accg1, false, false);
      accu0 = __builtin_amdgcn_wmma_f32_16x16x32_bf16(false, a0.v, false, b1.v, (short)0, accu0, false, false);
      accu1 = __builtin_amdgcn_wmma_f32_16x16x32_bf16(false, a1.v, false, b1.v, (short)0, accu1, false, false);
    }
  }
  int mofs = (lane & 16) ? 8 : 0;
  #pragma unroll
  for (int r = 0; r < 8; ++r) {
    float g0 = accg0[r], u0 = accu0[r];
    float h0 = g0 * (1.f / (1.f + __expf(-g0))) * u0;     // silu(g)*u, row group 0
    hb[(size_t)(row_base + r + mofs) * I_DIM + i_base + n] = f32_to_bf16(h0);
    float g1 = accg1[r], u1 = accu1[r];
    float h1 = g1 * (1.f / (1.f + __expf(-g1))) * u1;     // row group 1
    hb[(size_t)(row_base + 16 + r + mofs) * I_DIM + i_base + n] = f32_to_bf16(h1);
  }
}

// ---------------- down GEMM + weighted atomic scatter ----------------
// 32 tokens x 32 H-dims per wave; 4 WMMAs (2 row-groups x 2 N-tiles) per k-step.
#define DN_CHUNK  352    // 1408 / 4, = 11 k-steps of 32
#define DN_STRIDE 360
__global__ void __launch_bounds__(256) k_down(
    const unsigned short* __restrict__ hb,
    const unsigned short* __restrict__ dnb,
    const int* __restrict__ offs,
    const int* __restrict__ tok,
    const float* __restrict__ wrow,
    const int* __restrict__ erow,
    float* __restrict__ out) {
  __shared__ unsigned short sA[32 * DN_STRIDE];   // 23 KB
  int total = offs[E_NUM];
  int row_base = blockIdx.y * 32;
  if (row_base >= total) return;
  int e = erow[row_base];
  int tid = threadIdx.x;
  int wave = tid >> 5, lane = tid & 31;
  int hd0 = blockIdx.x * 256 + wave * 32;         // this wave's two N tiles
  int hd1 = hd0 + 16;
  int m   = lane & 15;
  int n   = lane & 15;
  int kof = (lane & 16) ? 8 : 0;
  const unsigned short* br0 = dnb + ((size_t)e * H_DIM + hd0 + n) * I_DIM;
  const unsigned short* br1 = dnb + ((size_t)e * H_DIM + hd1 + n) * I_DIM;
  v8f acc00 = {}, acc01 = {}, acc10 = {}, acc11 = {};

  for (int cc = 0; cc < I_DIM / DN_CHUNK; ++cc) {
    __syncthreads();
    // stage 32 x 352 bf16 = 5632 dwords, 22/thread
    #pragma unroll
    for (int it = 0; it < 22; ++it) {
      int idx = it * 256 + tid;
      int r = idx / 176;
      int c = idx % 176;
      unsigned int d = *(const unsigned int*)&hb[(size_t)(row_base + r) * I_DIM + cc * DN_CHUNK + c * 2];
      *(unsigned int*)&sA[r * DN_STRIDE + c * 2] = d;
    }
    __syncthreads();
    for (int kk = 0; kk < DN_CHUNK / 32; ++kk) {
      FragAB a0, a1, b0, b1;
      int kb_l = kk * 32;
      int kb_g = cc * DN_CHUNK + kk * 32;
      #pragma unroll
      for (int jj = 0; jj < 8; ++jj) {
        int k0 = kmap(jj, kof);
        a0.u[jj] = *(const unsigned int*)&sA[m * DN_STRIDE + kb_l + k0];
        a1.u[jj] = *(const unsigned int*)&sA[(m + 16) * DN_STRIDE + kb_l + k0];
        b0.u[jj] = *(const unsigned int*)&br0[kb_g + k0];
        b1.u[jj] = *(const unsigned int*)&br1[kb_g + k0];
      }
      acc00 = __builtin_amdgcn_wmma_f32_16x16x32_bf16(false, a0.v, false, b0.v, (short)0, acc00, false, false);
      acc01 = __builtin_amdgcn_wmma_f32_16x16x32_bf16(false, a0.v, false, b1.v, (short)0, acc01, false, false);
      acc10 = __builtin_amdgcn_wmma_f32_16x16x32_bf16(false, a1.v, false, b0.v, (short)0, acc10, false, false);
      acc11 = __builtin_amdgcn_wmma_f32_16x16x32_bf16(false, a1.v, false, b1.v, (short)0, acc11, false, false);
    }
  }
  int mofs = (lane & 16) ? 8 : 0;
  #pragma unroll
  for (int r = 0; r < 8; ++r) {
    {
      int row = row_base + r + mofs;               // row group 0
      int t = tok[row];
      float w = wrow[row];
      atomicAdd(&out[(size_t)t * H_DIM + hd0 + n], acc00[r] * w);
      atomicAdd(&out[(size_t)t * H_DIM + hd1 + n], acc01[r] * w);
    }
    {
      int row = row_base + 16 + r + mofs;          // row group 1
      int t = tok[row];
      float w = wrow[row];
      atomicAdd(&out[(size_t)t * H_DIM + hd0 + n], acc10[r] * w);
      atomicAdd(&out[(size_t)t * H_DIM + hd1 + n], acc11[r] * w);
    }
  }
}

// ---------------- host launcher ----------------
extern "C" void kernel_launch(void* const* d_in, const int* in_sizes, int n_in,
                              void* d_out, int out_size, void* d_ws, size_t ws_size,
                              hipStream_t stream) {
  (void)in_sizes; (void)n_in; (void)out_size; (void)ws_size;
  const float* x  = (const float*)d_in[0];   // [T, H]
  const float* wr = (const float*)d_in[1];   // [E, H]
  const float* gu = (const float*)d_in[2];   // [E, 2I, H]
  const float* dn = (const float*)d_in[3];   // [E, H, I]
  float* out = (float*)d_out;                // [T, H]
  char* ws = (char*)d_ws;

  int*   counts = (int*)(ws + OFF_COUNTS);
  int*   cursor = (int*)(ws + OFF_CURSOR);
  int*   offs   = (int*)(ws + OFF_OFFS);
  int*   topi   = (int*)(ws + OFF_TOPI);
  float* topw   = (float*)(ws + OFF_TOPW);
  int*   tok    = (int*)(ws + OFF_TOK);
  float* wrow   = (float*)(ws + OFF_WROW);
  int*   erow   = (int*)(ws + OFF_EROW);
  unsigned short* xb  = (unsigned short*)(ws + OFF_XB);
  unsigned short* gub = (unsigned short*)(ws + OFF_GUB);
  unsigned short* dnb = (unsigned short*)(ws + OFF_DNB);
  unsigned short* hb  = (unsigned short*)(ws + OFF_HB);

  k_init<<<1024, 256, 0, stream>>>(out, T_TOKENS * H_DIM, counts);
  k_cvt<<<2048, 256, 0, stream>>>(x,  xb,  T_TOKENS * H_DIM);
  k_cvt<<<2048, 256, 0, stream>>>(gu, gub, E_NUM * 2 * I_DIM * H_DIM);
  k_cvt<<<2048, 256, 0, stream>>>(dn, dnb, E_NUM * H_DIM * I_DIM);
  k_router<<<T_TOKENS / 8, 256, 0, stream>>>(x, wr, counts, topi, topw);
  k_scan<<<1, 32, 0, stream>>>(counts, offs, cursor);
  k_fill<<<(NROWS_MAX + 255) / 256, 256, 0, stream>>>(offs, tok, wrow, erow);
  k_build<<<(T_TOKENS + 255) / 256, 256, 0, stream>>>(topi, topw, cursor, tok, wrow);

  dim3 ggu(I_DIM / 128, ROW_TILES);   // 11 x 264
  k_gu<<<ggu, 256, 0, stream>>>(xb, gub, offs, tok, erow, hb);
  dim3 gdn(H_DIM / 256, ROW_TILES);   // 4 x 264
  k_down<<<gdn, 256, 0, stream>>>(hb, dnb, offs, tok, wrow, erow, out);
}
